// GNNEncoder_90890097918029
// MI455X (gfx1250) — matrix operand order repaired and checked by reading, not compiled, not run on hardware
//
#include <hip/hip_runtime.h>
#include <hip/hip_bf16.h>
#include <stdint.h>

typedef __bf16 bf16;
typedef __bf16 v16bf __attribute__((ext_vector_type(16)));
typedef __bf16 v8bf  __attribute__((ext_vector_type(8)));
typedef float  v8f   __attribute__((ext_vector_type(8)));

#define NN 50000
#define EE 800000
#define DD 128
#define HH 256
#define AA 16
#define WE1K 296   // 272 valid K, padded to 296 halfwords: 592B row = 148 DW = 20 mod 64 banks
                   // -> 16-lane x 16B fragment reads are LDS bank-conflict-free
#define TPW 5      // edge tiles per wave (amortize LDS weight staging)

static __device__ __forceinline__ v16bf make_frag(const bf16* p0, const bf16* p1) {
    union { v16bf v; v8bf h[2]; } u;
    u.h[0] = *reinterpret_cast<const v8bf*>(p0);
    u.h[1] = *reinterpret_cast<const v8bf*>(p1);
    return u.v;
}

static __device__ __forceinline__ v8f wmma_bf16(v16bf a, v16bf b, v8f c) {
    return __builtin_amdgcn_wmma_f32_16x16x32_bf16(false, a, false, b,
                                                   (short)0, c, false, false);
}

// ---- CDNA5 async global->LDS copy (ASYNCcnt path), 16B per lane ----
typedef __attribute__((address_space(3))) const char* lds_cptr_t;

static __device__ __forceinline__ unsigned lds_off(const void* p) {
    return (unsigned)(uintptr_t)(lds_cptr_t)(const char*)p;
}

static __device__ __forceinline__ void async_ld_b128(const void* gp, void* lp) {
    asm volatile("global_load_async_to_lds_b128 %0, %1, off"
                 :: "v"(lds_off(lp)), "v"((unsigned long long)(uintptr_t)gp)
                 : "memory");
}

static __device__ __forceinline__ void wait_asynccnt0() {
#if __has_builtin(__builtin_amdgcn_s_wait_asynccnt)
    __builtin_amdgcn_s_wait_asynccnt(0);
#else
    asm volatile("s_wait_asynccnt 0x0" ::: "memory");
#endif
}

// ---------------- elementwise helpers ----------------

__global__ void k_fill_f32(float* p, float v, int n) {
    int i = blockIdx.x * blockDim.x + threadIdx.x;
    if (i < n) p[i] = v;
}

__global__ void k_zero_f32(float* p, long long n) {
    long long i = (long long)blockIdx.x * blockDim.x + threadIdx.x;
    if (i < n) p[i] = 0.f;
}

__global__ void k_deg_count(const int* __restrict__ col, float* __restrict__ deg, int e) {
    int i = blockIdx.x * blockDim.x + threadIdx.x;
    if (i < e) atomicAdd(&deg[col[i]], 1.0f);
}

__global__ void k_dinv(const float* __restrict__ deg, float* __restrict__ dinv, int n) {
    int i = blockIdx.x * blockDim.x + threadIdx.x;
    if (i < n) dinv[i] = __frsqrt_rn(deg[i]);   // deg >= 1 always (self loops)
}

__global__ void k_f2bf(const float* __restrict__ s, bf16* __restrict__ d, long long n) {
    long long i = (long long)blockIdx.x * blockDim.x + threadIdx.x;
    if (i < n) d[i] = (bf16)s[i];
}

// W [K x Nn] row-major fp32  ->  Wt [Nn x strideK] bf16 (zero padded K..strideK)
__global__ void k_wt_bf(const float* __restrict__ W, bf16* __restrict__ Wt,
                        int K, int Nn, int strideK) {
    int i = blockIdx.x * blockDim.x + threadIdx.x;
    int total = Nn * strideK;
    if (i >= total) return;
    int n = i / strideK, k = i - n * strideK;
    Wt[i] = (k < K) ? (bf16)W[(size_t)k * Nn + n] : (bf16)0.0f;
}

// ---------------- generic bf16 WMMA GEMM ----------------
// C[M x Nn] = A[M x K](bf16) @ B (Bt[Nn x K] bf16)   (+bias, relu)
// grid (M/16, Nn/64), block 32 (one wave, 16x64 tile, 4 accumulators)

__global__ void gemm_bf16_wmma(const bf16* __restrict__ A, const bf16* __restrict__ Bt,
                               const float* __restrict__ bias, float* __restrict__ Cout,
                               bf16* __restrict__ Cbf, int Nn, int K, int relu) {
    const int lane = threadIdx.x & 31;
    const int m16  = lane & 15;
    const int half = lane >> 4;
    const int row0 = blockIdx.x * 16;
    const int col0 = blockIdx.y * 64;

    v8f acc[4];
#pragma unroll
    for (int t = 0; t < 4; ++t)
#pragma unroll
        for (int q = 0; q < 8; ++q) acc[t][q] = 0.f;

    const bf16* arow = A + (size_t)(row0 + m16) * K;
#pragma unroll 2
    for (int kb = 0; kb < K; kb += 32) {
        v16bf af = make_frag(arow + kb + 8 * half, arow + kb + 16 + 8 * half);
#pragma unroll
        for (int t = 0; t < 4; ++t) {
            int n = col0 + t * 16 + m16;
            const bf16* brow = Bt + (size_t)n * K + kb + 8 * half;
            v16bf bfr = make_frag(brow, brow + 16);
            acc[t] = wmma_bf16(af, bfr, acc[t]);
        }
    }
#pragma unroll
    for (int t = 0; t < 4; ++t) {
        int n = col0 + t * 16 + m16;
        float bv = bias ? bias[n] : 0.f;
#pragma unroll
        for (int v = 0; v < 8; ++v) {
            int mm = row0 + v + 8 * half;
            float val = acc[t][v] + bv;
            if (relu) val = fmaxf(val, 0.f);
            size_t o = (size_t)mm * Nn + n;
            if (Cout) Cout[o] = val;
            if (Cbf)  Cbf[o]  = (bf16)val;
        }
    }
}

// ---------------- GCN scatter-add aggregation ----------------
// acc[col[e]][c] += dinv[row[e]] * XW[row[e]][c]   (one wave per edge)

__global__ void k_scatter(const float* __restrict__ XW, const float* __restrict__ dinv,
                          const int* __restrict__ row, const int* __restrict__ col,
                          float* __restrict__ acc, int e, int C) {
    int eid = blockIdx.x * 8 + (threadIdx.x >> 5);
    if (eid >= e) return;
    int lane = threadIdx.x & 31;
    int r = row[eid], c = col[eid];
    float w = dinv[r];
    const float* src = XW + (size_t)r * C;
    float*       dst = acc + (size_t)c * C;
    for (int ch = lane; ch < C; ch += 32)
        atomicAdd(&dst[ch], w * src[ch]);
}

// h[i][c] = relu( dinv[i]*(acc[i][c] + dinv[i]*XW[i][c]) + b[c] )  -> bf16
__global__ void k_finalize(const float* __restrict__ acc, const float* __restrict__ XW,
                           const float* __restrict__ dinv, const float* __restrict__ bias,
                           bf16* __restrict__ hbf, int C, long long total) {
    long long i = (long long)blockIdx.x * blockDim.x + threadIdx.x;
    if (i >= total) return;
    int node = (int)(i / C);
    int ch   = (int)(i - (long long)node * C);
    float di = dinv[node];
    float v  = di * (acc[i] + di * XW[i]) + bias[ch];
    hbf[i] = (bf16)fmaxf(v, 0.f);
}

// ---------------- fused edge-predictor MLP ----------------
// Block = 8 waves. The whole We1t panel [256 x 296] bf16 (148KB) is staged in
// LDS ONCE per block via async global->LDS copies (ASYNCcnt), then each wave
// processes TPW=5 tiles of 16 edges: hidden[16x256] is computed 32 cols at a
// time with WMMA (B-frags from LDS), bias+relu'd, re-laid-out through a small
// per-wave LDS tile (D-layout -> A-layout) and folded into out += hid @ We2.
// grid = EE/16/(8*TPW) = 1250.

__global__ void edge_mlp_wmma(const bf16* __restrict__ Hbf, const int* __restrict__ rowIdx,
                              const int* __restrict__ colIdx, const float* __restrict__ eattr,
                              const bf16* __restrict__ We1t, const float* __restrict__ be1,
                              const bf16* __restrict__ We2t, const float* __restrict__ be2,
                              float* __restrict__ out) {
    extern __shared__ char smem_raw[];
    bf16* weS   = (bf16*)smem_raw;                                   // [HH][WE1K]
    bf16* stage = (bf16*)(smem_raw + (size_t)HH * WE1K * 2);         // [8][16*32]

    const int wave = threadIdx.x >> 5;
    const int lane = threadIdx.x & 31;
    const int m16  = lane & 15;
    const int half = lane >> 4;

    // ---- cooperative async stage of We1t into LDS (9472 x 16B chunks) ----
    {
        const char* g = (const char*)We1t;
        char*       l = (char*)weS;
        const int nchunk = (HH * WE1K * 2) / 16;   // 9472
        for (int i = threadIdx.x; i < nchunk; i += 256)
            async_ld_b128(g + 16 * i, l + 16 * i);
        wait_asynccnt0();
    }
    __syncthreads();

    bf16* myStage = stage + wave * (16 * 32);

    for (int tt = 0; tt < TPW; ++tt) {
        const int tile = (blockIdx.x * 8 + wave) * TPW + tt;
        const int e0   = tile * 16;

        const int er = rowIdx[e0 + m16];
        const int ec = colIdx[e0 + m16];

        // A fragments, kept in registers for the whole tile
        v16bf aR[4], aC[4], aT;
        const bf16* hr = Hbf + (size_t)er * DD;
        const bf16* hc = Hbf + (size_t)ec * DD;
#pragma unroll
        for (int kc = 0; kc < 4; ++kc) {
            aR[kc] = make_frag(hr + kc * 32 + 8 * half, hr + kc * 32 + 16 + 8 * half);
            aC[kc] = make_frag(hc + kc * 32 + 8 * half, hc + kc * 32 + 16 + 8 * half);
        }
        {
            union { v16bf v; v8bf h[2]; } u;
            const float* ap = eattr + (size_t)(e0 + m16) * AA + 8 * half;
#pragma unroll
            for (int q = 0; q < 8; ++q) u.h[0][q] = (bf16)ap[q];
#pragma unroll
            for (int q = 0; q < 8; ++q) u.h[1][q] = (bf16)0.f;
            aT = u.v;
        }

        v8f oacc;
#pragma unroll
        for (int q = 0; q < 8; ++q) oacc[q] = 0.f;

        for (int j = 0; j < 8; ++j) {
            const int jb = j * 32;
            v8f h0, h1;
#pragma unroll
            for (int q = 0; q < 8; ++q) { h0[q] = 0.f; h1[q] = 0.f; }

            const bf16* wr0 = weS + (size_t)(jb + m16) * WE1K;        // LDS, conflict-free
            const bf16* wr1 = weS + (size_t)(jb + 16 + m16) * WE1K;
#pragma unroll
            for (int kc = 0; kc < 4; ++kc) {
                const bf16* b0 = wr0 + kc * 32 + 8 * half;            // h[row] block, K 0..127
                const bf16* b1 = wr1 + kc * 32 + 8 * half;
                h0 = wmma_bf16(aR[kc], make_frag(b0, b0 + 16), h0);
                h1 = wmma_bf16(aR[kc], make_frag(b1, b1 + 16), h1);
                const bf16* c0 = wr0 + 128 + kc * 32 + 8 * half;      // h[col] block, K 128..255
                const bf16* c1 = wr1 + 128 + kc * 32 + 8 * half;
                h0 = wmma_bf16(aC[kc], make_frag(c0, c0 + 16), h0);
                h1 = wmma_bf16(aC[kc], make_frag(c1, c1 + 16), h1);
            }
            {   // edge_attr block, K 256..271 (272..295 are zero pad)
                const bf16* t0 = wr0 + 256 + 8 * half;
                const bf16* t1 = wr1 + 256 + 8 * half;
                h0 = wmma_bf16(aT, make_frag(t0, t0 + 16), h0);
                h1 = wmma_bf16(aT, make_frag(t1, t1 + 16), h1);
            }

            // bias + relu, stage 16x32 hidden tile (D-layout -> A-layout)
            const float bb0 = be1[jb + m16];
            const float bb1 = be1[jb + 16 + m16];
#pragma unroll
            for (int v = 0; v < 8; ++v) {
                int mm = v + 8 * half;
                myStage[mm * 32 + m16]      = (bf16)fmaxf(h0[v] + bb0, 0.f);
                myStage[mm * 32 + 16 + m16] = (bf16)fmaxf(h1[v] + bb1, 0.f);
            }
            __syncthreads();
            v16bf a2 = make_frag(myStage + m16 * 32 + 8 * half,
                                 myStage + m16 * 32 + 16 + 8 * half);
            const bf16* b2p = We2t + (size_t)m16 * HH + jb + 8 * half;
            oacc = wmma_bf16(a2, make_frag(b2p, b2p + 16), oacc);
            __syncthreads();
        }

        const float bv = be2[m16];
#pragma unroll
        for (int v = 0; v < 8; ++v) {
            int mm = v + 8 * half;
            out[(size_t)(e0 + mm) * AA + m16] = oacc[v] + bv;
        }
    }
}

// ---------------- host orchestration ----------------

static inline void* wsalloc(char*& cur, size_t bytes) {
    void* p = (void*)cur;
    cur += (bytes + 255) & ~(size_t)255;
    return p;
}

extern "C" void kernel_launch(void* const* d_in, const int* in_sizes, int n_in,
                              void* d_out, int out_size, void* d_ws, size_t ws_size,
                              hipStream_t stream) {
    (void)in_sizes; (void)n_in; (void)out_size; (void)ws_size;

    const float* x    = (const float*)d_in[0];
    const int*   eidx = (const int*)d_in[1];
    const float* eatt = (const float*)d_in[2];
    const float* W1   = (const float*)d_in[3];
    const float* b1   = (const float*)d_in[4];
    const float* W2   = (const float*)d_in[5];
    const float* b2   = (const float*)d_in[6];
    const float* Wo1  = (const float*)d_in[7];
    const float* bo1  = (const float*)d_in[8];
    const float* Wo2  = (const float*)d_in[9];
    const float* bo2  = (const float*)d_in[10];
    const float* We1  = (const float*)d_in[11];
    const float* be1  = (const float*)d_in[12];
    const float* We2  = (const float*)d_in[13];
    const float* be2  = (const float*)d_in[14];

    const int* rowI = eidx;
    const int* colI = eidx + EE;

    char* cur = (char*)d_ws;
    float* deg   = (float*)wsalloc(cur, NN * 4);
    float* dinv  = (float*)wsalloc(cur, NN * 4);
    bf16*  xbf   = (bf16*) wsalloc(cur, (size_t)NN * DD * 2);  // reused as final-h bf16
    bf16*  W1t   = (bf16*) wsalloc(cur, (size_t)HH * DD * 2);
    bf16*  W2t   = (bf16*) wsalloc(cur, (size_t)DD * HH * 2);
    bf16*  Wo1t  = (bf16*) wsalloc(cur, (size_t)HH * DD * 2);
    bf16*  Wo2t  = (bf16*) wsalloc(cur, (size_t)DD * HH * 2);
    bf16*  We1t  = (bf16*) wsalloc(cur, (size_t)HH * WE1K * 2);
    bf16*  We2t  = (bf16*) wsalloc(cur, (size_t)AA * HH * 2);
    float* XW    = (float*)wsalloc(cur, (size_t)NN * HH * 4);   // GEMM out (max H)
    float* accB  = (float*)wsalloc(cur, (size_t)NN * HH * 4);   // scatter accumulator
    bf16*  hbuf  = (bf16*) wsalloc(cur, (size_t)NN * HH * 2);   // h1 / h2 bf16
    bf16*  Tbf   = (bf16*) wsalloc(cur, (size_t)NN * HH * 2);   // MLP hidden bf16

    float* hout  = (float*)d_out;                    // [NN x DD]
    float* eaout = (float*)d_out + (size_t)NN * DD;  // [EE x AA]

    const int TB = 256;
    auto blks = [](long long n, int tb) { return (unsigned)((n + tb - 1) / tb); };

    // --- degree / normalization ---
    k_fill_f32<<<blks(NN, TB), TB, 0, stream>>>(deg, 1.0f, NN);  // self loops
    k_deg_count<<<blks(EE, TB), TB, 0, stream>>>(colI, deg, EE);
    k_dinv<<<blks(NN, TB), TB, 0, stream>>>(deg, dinv, NN);

    // --- precision conversion / weight transposition ---
    k_f2bf<<<blks((long long)NN * DD, TB), TB, 0, stream>>>(x, xbf, (long long)NN * DD);
    k_wt_bf<<<blks(HH * DD, TB), TB, 0, stream>>>(W1,  W1t,  DD,  HH, DD);   // W1 [D,H] -> [H][D]
    k_wt_bf<<<blks(DD * HH, TB), TB, 0, stream>>>(W2,  W2t,  HH,  DD, HH);   // W2 [H,D] -> [D][H]
    k_wt_bf<<<blks(HH * DD, TB), TB, 0, stream>>>(Wo1, Wo1t, DD,  HH, DD);
    k_wt_bf<<<blks(DD * HH, TB), TB, 0, stream>>>(Wo2, Wo2t, HH,  DD, HH);
    k_wt_bf<<<blks(HH * WE1K, TB), TB, 0, stream>>>(We1, We1t, 2 * DD + AA, HH, WE1K);
    k_wt_bf<<<blks(AA * HH, TB), TB, 0, stream>>>(We2, We2t, HH,  AA, HH);   // We2 [H,A] -> [A][H]

    // --- GCN layer 1: XW = x @ W1 ; aggregate ; h1 = relu(...) ---
    gemm_bf16_wmma<<<dim3(NN / 16, HH / 64), 32, 0, stream>>>(
        xbf, W1t, nullptr, XW, nullptr, HH, DD, 0);
    k_zero_f32<<<blks((long long)NN * HH, TB), TB, 0, stream>>>(accB, (long long)NN * HH);
    k_scatter<<<blks((long long)EE * 32, TB), TB, 0, stream>>>(XW, dinv, rowI, colI, accB, EE, HH);
    k_finalize<<<blks((long long)NN * HH, TB), TB, 0, stream>>>(
        accB, XW, dinv, b1, hbuf, HH, (long long)NN * HH);

    // --- GCN layer 2: XW2 = h1 @ W2 ; aggregate ; h2 = relu(...) ---
    gemm_bf16_wmma<<<dim3(NN / 16, DD / 64), 32, 0, stream>>>(
        hbuf, W2t, nullptr, XW, nullptr, DD, HH, 0);
    k_zero_f32<<<blks((long long)NN * DD, TB), TB, 0, stream>>>(accB, (long long)NN * DD);
    k_scatter<<<blks((long long)EE * 32, TB), TB, 0, stream>>>(XW, dinv, rowI, colI, accB, EE, DD);
    k_finalize<<<blks((long long)NN * DD, TB), TB, 0, stream>>>(
        accB, XW, dinv, b2, hbuf, DD, (long long)NN * DD);   // hbuf now h2 [NN x DD] bf16

    // --- node MLP: T = relu(h2 @ Wo1 + bo1) ; h = T @ Wo2 + bo2 ---
    gemm_bf16_wmma<<<dim3(NN / 16, HH / 64), 32, 0, stream>>>(
        hbuf, Wo1t, bo1, nullptr, Tbf, HH, DD, 1);
    gemm_bf16_wmma<<<dim3(NN / 16, DD / 64), 32, 0, stream>>>(
        Tbf, Wo2t, bo2, hout, xbf /* h bf16 for edge gather */, DD, HH, 0);

    // --- fused edge predictor: LDS-staged weights, 640 edges/block ---
    const size_t edgeSmem = (size_t)HH * WE1K * 2 + 8 * 16 * 32 * 2;  // ~156 KB
    edge_mlp_wmma<<<EE / (16 * 8 * TPW), 256, edgeSmem, stream>>>(
        xbf, rowI, colI, eatt, We1t, be1, We2t, be2, eaout);
}
